// ChannelGuidedAttn_35579509080688
// MI455X (gfx1250) — matrix operand order repaired and checked by reading, not compiled, not run on hardware
//
#include <hip/hip_runtime.h>
#include <hip/hip_bf16.h>
#include <math.h>

typedef __attribute__((ext_vector_type(16))) __bf16 v16bf;
typedef __attribute__((ext_vector_type(8)))  __bf16 v8bf;
typedef __attribute__((ext_vector_type(4)))  __bf16 v4bf;
typedef __attribute__((ext_vector_type(8)))  float  v8f;
typedef __attribute__((ext_vector_type(4)))  float  v4f;

#define BB 16
#define CC 512
#define NN 4096   // H*W
#define WW 64

__device__ __forceinline__ v16bf cat16(v8bf lo, v8bf hi) {
  return __builtin_shufflevector(lo, hi, 0,1,2,3,4,5,6,7,8,9,10,11,12,13,14,15);
}

__device__ __forceinline__ v8f wmma_bf16(v16bf a, v16bf b, v8f c) {
  // D = A*B + C, f32 accumulate, 16x16x32 bf16
  return __builtin_amdgcn_wmma_f32_16x16x32_bf16(false, a, false, b, (short)0, c, false, false);
}

// ---------------------------------------------------------------------------
// Kernel 0a: fp32 -> bf16 elementwise (4 elems / thread, b128 in, b64 out)
// ---------------------------------------------------------------------------
__global__ __launch_bounds__(256) void cvt_bf16_kernel(const float* __restrict__ src,
                                                       __bf16* __restrict__ dst, int n4) {
  int i = blockIdx.x * blockDim.x + threadIdx.x;
  if (i < n4) {
    v4f v = *(const v4f*)(src + (size_t)i * 4);
    v4bf o;
    o.x = (__bf16)v.x; o.y = (__bf16)v.y; o.z = (__bf16)v.z; o.w = (__bf16)v.w;
    *(v4bf*)(dst + (size_t)i * 4) = o;
  }
}

// ---------------------------------------------------------------------------
// Kernel 0b: x_training [B][C][N] f32 -> kvT [B][N][C] bf16 (LDS 32x32 tiles)
// ---------------------------------------------------------------------------
__global__ __launch_bounds__(256) void transpose_cvt_kernel(const float* __restrict__ x,
                                                            __bf16* __restrict__ xT) {
  __shared__ float tile[32][33];
  int b  = blockIdx.z;
  int n0 = blockIdx.x * 32;
  int c0 = blockIdx.y * 32;
  int tx = threadIdx.x;        // 0..31
  int ty = threadIdx.y;        // 0..7
#pragma unroll
  for (int i = 0; i < 32; i += 8) {
    tile[ty + i][tx] = x[((size_t)b * CC + (c0 + ty + i)) * NN + n0 + tx];
  }
  __syncthreads();
#pragma unroll
  for (int i = 0; i < 32; i += 8) {
    xT[((size_t)b * NN + (n0 + ty + i)) * CC + c0 + tx] = (__bf16)tile[tx][ty + i];
  }
}

// ---------------------------------------------------------------------------
// Kernel 1: energy[b,c,d] = sum_n qb[b,c,n]*kvb[b,d,n]
// One wave computes a 32(c) x 64(d) tile: 2 A frags x 4 B frags = 8 WMMA/k-step.
// B frags are reused across both A row-groups (12 b128 loads -> 8 WMMAs).
// ---------------------------------------------------------------------------
__global__ __launch_bounds__(32) void energy_gemm_kernel(const __bf16* __restrict__ qb,
                                                         const __bf16* __restrict__ kvb,
                                                         float* __restrict__ energy) {
  int b    = blockIdx.z;
  int c0   = blockIdx.y * 32;
  int d0   = blockIdx.x * 64;
  int lane = threadIdx.x;
  int half = lane >> 4;        // 0 | 1
  int l15  = lane & 15;

  int aoff = half * 8;         // A frag: K chunks {0..7,16..23} or {8..15,24..31}
  const __bf16* qrow[2];
#pragma unroll
  for (int g = 0; g < 2; ++g)
    qrow[g] = qb + ((size_t)b * CC + (c0 + 16 * g + l15)) * NN + aoff;

  const __bf16* krow[4];
#pragma unroll
  for (int t = 0; t < 4; ++t)  // B frag: lane = column d, K chunks {0..7,16..23} or {8..15,24..31}
    krow[t] = kvb + ((size_t)b * CC + (d0 + 16 * t + l15)) * NN + half * 8;

  v8f acc[2][4];
#pragma unroll
  for (int g = 0; g < 2; ++g)
#pragma unroll
    for (int t = 0; t < 4; ++t) acc[g][t] = v8f{0.f};

  for (int k = 0; k < NN; k += 32) {
    v16bf a[2];
#pragma unroll
    for (int g = 0; g < 2; ++g)
      a[g] = cat16(*(const v8bf*)(qrow[g] + k), *(const v8bf*)(qrow[g] + k + 16));
#pragma unroll
    for (int t = 0; t < 4; ++t) {
      v16bf bm = cat16(*(const v8bf*)(krow[t] + k), *(const v8bf*)(krow[t] + k + 16));
      acc[0][t] = wmma_bf16(a[0], bm, acc[0][t]);
      acc[1][t] = wmma_bf16(a[1], bm, acc[1][t]);
    }
  }

  int m0 = half * 8;           // D layout: VGPR v -> row m0+v, col = l15
#pragma unroll
  for (int g = 0; g < 2; ++g)
#pragma unroll
    for (int t = 0; t < 4; ++t)
#pragma unroll
      for (int v = 0; v < 8; ++v)
        energy[((size_t)b * CC + (c0 + 16 * g + m0 + v)) * CC + d0 + 16 * t + l15] =
            acc[g][t][v];
}

// ---------------------------------------------------------------------------
// Kernel 2: attention row softmax.  softmax(max_e - e) == exp(min_e - e)/sum
// One 128-thread block per (b,c) row of 512; writes bf16.
// ---------------------------------------------------------------------------
__global__ __launch_bounds__(128) void attn_softmax_kernel(const float* __restrict__ energy,
                                                           __bf16* __restrict__ attn) {
  __shared__ float red[8];
  size_t row = blockIdx.x;
  const float* e = energy + row * CC;
  __bf16* a = attn + row * CC;
  int t = threadIdx.x, wid = t >> 5;

  float v[4], mn = 3.0e38f;
#pragma unroll
  for (int i = 0; i < 4; ++i) { v[i] = e[t + 128 * i]; mn = fminf(mn, v[i]); }
#pragma unroll
  for (int m = 16; m >= 1; m >>= 1) mn = fminf(mn, __shfl_xor(mn, m, 32));
  if ((t & 31) == 0) red[wid] = mn;
  __syncthreads();
  mn = fminf(fminf(red[0], red[1]), fminf(red[2], red[3]));

  float p[4], s = 0.f;
#pragma unroll
  for (int i = 0; i < 4; ++i) { p[i] = __expf(mn - v[i]); s += p[i]; }
#pragma unroll
  for (int m = 16; m >= 1; m >>= 1) s += __shfl_xor(s, m, 32);
  if ((t & 31) == 0) red[4 + wid] = s;
  __syncthreads();
  s = (red[4] + red[5]) + (red[6] + red[7]);

  float inv = 1.0f / s;
#pragma unroll
  for (int i = 0; i < 4; ++i) a[t + 128 * i] = (__bf16)(p[i] * inv);
}

// ---------------------------------------------------------------------------
// Kernel 3: out[b,c,n] = sum_d attn[b,c,d]*kvT[b,n,d], then softmax over w=n%64
// One wave per 32(c) x 64(n) tile (spans full W rows), fused width-softmax
// epilogue via half-wave shuffles (lanes 0-15 hold row m=v, 16-31 row m=v+8).
// ---------------------------------------------------------------------------
__global__ __launch_bounds__(32) void out_gemm_kernel(const __bf16* __restrict__ attn,
                                                      const __bf16* __restrict__ kvT,
                                                      float* __restrict__ out) {
  int b    = blockIdx.z;
  int c0   = blockIdx.y * 32;
  int n0   = blockIdx.x * 64;
  int lane = threadIdx.x;
  int half = lane >> 4;
  int l15  = lane & 15;

  int aoff = half * 8;
  const __bf16* arow[2];
#pragma unroll
  for (int g = 0; g < 2; ++g)
    arow[g] = attn + ((size_t)b * CC + (c0 + 16 * g + l15)) * CC + aoff;

  const __bf16* brow[4];
#pragma unroll
  for (int t = 0; t < 4; ++t)
    brow[t] = kvT + ((size_t)b * NN + (n0 + 16 * t + l15)) * CC + half * 8;

  v8f acc[2][4];
#pragma unroll
  for (int g = 0; g < 2; ++g)
#pragma unroll
    for (int t = 0; t < 4; ++t) acc[g][t] = v8f{0.f};

  for (int k = 0; k < CC; k += 32) {
    v16bf a[2];
#pragma unroll
    for (int g = 0; g < 2; ++g)
      a[g] = cat16(*(const v8bf*)(arow[g] + k), *(const v8bf*)(arow[g] + k + 16));
#pragma unroll
    for (int t = 0; t < 4; ++t) {
      v16bf bm = cat16(*(const v8bf*)(brow[t] + k), *(const v8bf*)(brow[t] + k + 16));
      acc[0][t] = wmma_bf16(a[0], bm, acc[0][t]);
      acc[1][t] = wmma_bf16(a[1], bm, acc[1][t]);
    }
  }

  // Epilogue: per output row m = v + 8*half (+16 per row-group g), softmax over
  // the 64 w-values = {4 tiles t} x {16 lanes l15}.
  int m0 = half * 8;
#pragma unroll
  for (int g = 0; g < 2; ++g) {
#pragma unroll
    for (int v = 0; v < 8; ++v) {
      float mx = fmaxf(fmaxf(acc[g][0][v], acc[g][1][v]),
                       fmaxf(acc[g][2][v], acc[g][3][v]));
#pragma unroll
      for (int m = 8; m >= 1; m >>= 1) mx = fmaxf(mx, __shfl_xor(mx, m, 16));
      float p[4], s = 0.f;
#pragma unroll
      for (int t = 0; t < 4; ++t) { p[t] = __expf(acc[g][t][v] - mx); s += p[t]; }
#pragma unroll
      for (int m = 8; m >= 1; m >>= 1) s += __shfl_xor(s, m, 16);
      float inv = 1.0f / s;
#pragma unroll
      for (int t = 0; t < 4; ++t)
        out[((size_t)b * CC + (c0 + 16 * g + m0 + v)) * NN + n0 + 16 * t + l15] =
            p[t] * inv;
    }
  }
}

// ---------------------------------------------------------------------------
extern "C" void kernel_launch(void* const* d_in, const int* in_sizes, int n_in,
                              void* d_out, int out_size, void* d_ws, size_t ws_size,
                              hipStream_t stream) {
  const float* x_training = (const float*)d_in[0];  // kv
  const float* x_pre      = (const float*)d_in[1];  // q
  float* out = (float*)d_out;

  const size_t elems = (size_t)BB * CC * NN;        // 33,554,432
  char* ws = (char*)d_ws;
  __bf16* qb     = (__bf16*)(ws);                                   // 64 MiB
  __bf16* kvb    = (__bf16*)(ws + elems * 2);                       // 64 MiB
  __bf16* kvT    = (__bf16*)(ws + elems * 4);                       // 64 MiB
  float*  energy = (float*) (ws + elems * 6);                       // 16 MiB
  __bf16* attn   = (__bf16*)(ws + elems * 6 + (size_t)BB*CC*CC*4);  //  8 MiB

  // 0a. fp32 -> bf16 row-major copies
  int n4 = (int)(elems / 4);
  dim3 cgrid((n4 + 255) / 256), cblk(256);
  cvt_bf16_kernel<<<cgrid, cblk, 0, stream>>>(x_pre,      qb,  n4);
  cvt_bf16_kernel<<<cgrid, cblk, 0, stream>>>(x_training, kvb, n4);

  // 0b. fp32 -> bf16 transposed copy [B][N][C]
  transpose_cvt_kernel<<<dim3(NN / 32, CC / 32, BB), dim3(32, 8), 0, stream>>>(x_training, kvT);

  // 1. energy = q . kv^T   (B x C x C), 32x64 tiles per wave
  energy_gemm_kernel<<<dim3(CC / 64, CC / 32, BB), dim3(32), 0, stream>>>(qb, kvb, energy);

  // 2. attention = softmax(max - energy) per row
  attn_softmax_kernel<<<dim3(BB * CC), dim3(128), 0, stream>>>(energy, attn);

  // 3. out = attention . kv, fused final softmax over W, 32x64 tiles per wave
  out_gemm_kernel<<<dim3(NN / 64, CC / 32, BB), dim3(32), 0, stream>>>(attn, kvT, out);
}
